// SimilarityCorrector_58007828300453
// MI455X (gfx1250) — matrix operand order repaired and checked by reading, not compiled, not run on hardware
//
#include <hip/hip_runtime.h>
#include <cstdint>

// ---------------------------------------------------------------------------
// SimilarityCorrector for MI455X (gfx1250, wave32).
//
// Per pair (b,i,j): out = sigmoid( relu( relu(x*w1+b1) @ W2 + b2 ) @ w3 + b3 )
// then pair-mask, 0.5*(out+out^T), zero diagonal.
//
// Layer 2 (2048 FLOP/pair, dominant) runs on V_WMMA_F32_16X16X32_F16 as
//   h2^T = W2^T (A, loop-invariant) x h1^T (B, pairs on lane dim), C = b2.
// Layer 3 folds into a per-lane dot over the 8 D VGPRs + one xor-16 shuffle.
// j-loop unrolled x2 to hide WMMA->VALU hazard slots; 4-way split accumulator
// breaks the layer-3 fmac dependency chain.
// ---------------------------------------------------------------------------

typedef _Float16 __attribute__((ext_vector_type(16))) v16h;
typedef float    __attribute__((ext_vector_type(8)))  v8f;

#define B_ 8
#define N_ 1024
#define H_ 32

__global__ __launch_bounds__(128) void simcorr_mlp_kernel(
    const float*   __restrict__ sim,   // [B,N,N]
    const uint8_t* __restrict__ mask,  // [B,N]  (jax bool = 1 byte)
    const float*   __restrict__ w1,    // [1,H]
    const float*   __restrict__ b1,    // [H]
    const float*   __restrict__ w2,    // [H,H] row-major: w2[h_in*H + k_out]
    const float*   __restrict__ b2,    // [H]
    const float*   __restrict__ w3,    // [H,1]
    const float*   __restrict__ b3,    // [1]
    float*         __restrict__ out)   // [B,N,N] raw (masked, unsymmetrized)
{
  const int lane = threadIdx.x & 31;
  const int wid  = blockIdx.x * (blockDim.x >> 5) + (threadIdx.x >> 5); // 0..B*N-1
  const int b    = wid >> 10;          // / N_
  const int i    = wid & (N_ - 1);
  const int m    = lane & 15;          // row within A / pair within B / N within D
  const int hi   = lane >> 4;          // half-wave select

  // ---- loop-invariant per-lane constants --------------------------------
  // A-layout (16-bit A 16x32): lane holds row M=lane%16; VGPR v half h ->
  //   K = 16*(v>=4) + 8*hi + 2*(v&3) + h
  // B-layout (16-bit B 32x16): lane holds col N=lane%16; element e=2v+h ->
  //   K = 16*(e>=8) + 8*hi + (e&7)
  v16h a0 = {}, a1 = {}, w1v = {}, b1v = {};
  v8f  cc0 = {}, cc1 = {};
  float w3c0[8], w3c1[8];
#pragma unroll
  for (int v = 0; v < 8; ++v) {
#pragma unroll
    for (int h = 0; h < 2; ++h) {
      const int e  = 2 * v + h;
      const int ka = ((v & 4) << 2) + (hi << 3) + ((v & 3) << 1) + h;
      a0[e]  = (_Float16)w2[ka * H_ + m];        // A = W2^T, out-feats 0..15
      a1[e]  = (_Float16)w2[ka * H_ + 16 + m];   // A = W2^T, out-feats 16..31
      const int kb = ((e & 8) << 1) + (hi << 3) + (e & 7);
      w1v[e] = (_Float16)w1[kb];
      b1v[e] = (_Float16)b1[kb];
    }
    cc0[v]  = b2[v + 8 * hi];          // C/D layout: VGPR v, half hi -> M = v+8*hi
    cc1[v]  = b2[16 + v + 8 * hi];
    w3c0[v] = w3[v + 8 * hi];
    w3c1[v] = w3[16 + v + 8 * hi];
  }
  const float   b3s = b3[0];
  const uint8_t mi  = mask[b * N_ + i];
  const float*  row  = sim + ((size_t)(b * N_ + i)) * N_;
  float*        orow = out + ((size_t)(b * N_ + i)) * N_;

  // ---- loop over 64 j-tiles of 16 pairs; unroll x2 for latency hiding ----
#pragma unroll 2
  for (int j0 = 0; j0 < N_; j0 += 16) {
    const int   jj = j0 + m;
    const float x  = row[jj];                       // pair scalar input

    // layer 1 in packed f16: h1^T column for this lane (B matrix)
    const _Float16 xh = (_Float16)x;
    v16h xs;
#pragma unroll
    for (int e = 0; e < 16; ++e) xs[e] = xh;
    v16h bm = __builtin_elementwise_fma(xs, w1v, b1v);
    bm = __builtin_elementwise_max(bm, (v16h){});   // relu -> v_pk_max

    // layer 2: two 16x16x32 f16 WMMAs, bias via C
    v8f d0 = __builtin_amdgcn_wmma_f32_16x16x32_f16(
        false, a0, false, bm, (short)0, cc0, false, false);
    v8f d1 = __builtin_amdgcn_wmma_f32_16x16x32_f16(
        false, a1, false, bm, (short)0, cc1, false, false);

    // layer 3: per-lane dot over the 8 D VGPRs (feature dim), relu fused.
    // 4 independent accumulators -> short dependency chains, dual-issue.
    float r0 = 0.0f, r1 = 0.0f, r2 = 0.0f, r3 = 0.0f;
#pragma unroll
    for (int v = 0; v < 8; v += 2) {
      r0 = fmaf(fmaxf(d0[v],     0.0f), w3c0[v],     r0);
      r1 = fmaf(fmaxf(d0[v + 1], 0.0f), w3c0[v + 1], r1);
      r2 = fmaf(fmaxf(d1[v],     0.0f), w3c1[v],     r2);
      r3 = fmaf(fmaxf(d1[v + 1], 0.0f), w3c1[v + 1], r3);
    }
    float r = (r0 + r1) + (r2 + r3);
    // merge the two half-wave feature groups (each lane pair shares N=pair)
    r += __shfl_xor(r, 16, 32);

    float val = 1.0f / (1.0f + __expf(-(r + b3s)));
    const uint8_t mj = mask[b * N_ + jj];
    val = (mi && mj) ? val : 0.0f;

    if (lane < 16) orow[jj] = val;                  // 16 contiguous floats
  }
}

// In-place symmetrize + zero diagonal. Race-free: the (i<j) thread owns both
// (i,j) and (j,i); diagonal threads own (i,i). Transposed reads hit L2 (192MB).
__global__ __launch_bounds__(256) void simcorr_sym_kernel(float* __restrict__ out)
{
  const size_t idx = (size_t)blockIdx.x * blockDim.x + threadIdx.x; // B*N*N
  const int j = (int)(idx & (N_ - 1));
  const int i = (int)((idx >> 10) & (N_ - 1));
  if (i == j) { out[idx] = 0.0f; return; }
  if (i < j) {
    const size_t bbase = (idx >> 20) * (size_t)(N_ * N_);
    const size_t t = bbase + (size_t)j * N_ + i;
    const float a = out[idx];
    const float c = out[t];
    const float mv = 0.5f * (a + c);
    out[idx] = mv;
    out[t]   = mv;
  }
}

extern "C" void kernel_launch(void* const* d_in, const int* in_sizes, int n_in,
                              void* d_out, int out_size, void* d_ws, size_t ws_size,
                              hipStream_t stream)
{
  (void)in_sizes; (void)n_in; (void)out_size; (void)d_ws; (void)ws_size;

  const float*   sim  = (const float*)  d_in[0];
  const uint8_t* mask = (const uint8_t*)d_in[1];
  const float*   w1   = (const float*)  d_in[2];
  const float*   b1   = (const float*)  d_in[3];
  const float*   w2   = (const float*)  d_in[4];
  const float*   b2   = (const float*)  d_in[5];
  const float*   w3   = (const float*)  d_in[6];
  const float*   b3   = (const float*)  d_in[7];
  float*         out  = (float*)d_out;

  // One wave per (b,i) row: 8192 waves, 4 waves (128 thr) per block.
  dim3 blk1(128);
  dim3 grd1((B_ * N_) / 4);
  simcorr_mlp_kernel<<<grd1, blk1, 0, stream>>>(sim, mask, w1, b1, w2, b2, w3, b3, out);

  dim3 blk2(256);
  dim3 grd2((B_ * N_ * N_) / 256);
  simcorr_sym_kernel<<<grd2, blk2, 0, stream>>>(out);
}
